// meshgrapnent_38766374814183
// MI455X (gfx1250) — matrix operand order repaired and checked by reading, not compiled, not run on hardware
//
#include <hip/hip_runtime.h>
#include <hip/hip_bf16.h>

// ---------- WMMA types ----------
typedef __attribute__((ext_vector_type(16))) __bf16          v16bf;
typedef __attribute__((ext_vector_type(8)))  float           v8f;
typedef __attribute__((ext_vector_type(8)))  unsigned short  u16x8;
typedef __attribute__((ext_vector_type(16))) unsigned short  u16x16;

union BFrag {
  v16bf  bf;
  u16x8  h[2];
  u16x16 w;
};

__device__ __forceinline__ unsigned short f32_bf16(float f) {
  union { float f; unsigned u; } v; v.f = f;
  unsigned u = v.u;
  u += 0x7fffu + ((u >> 16) & 1u);          // round-to-nearest-even
  return (unsigned short)(u >> 16);
}

constexpr int D_   = 128;
constexpr int MT   = 64;        // rows per block (4 M-tiles of 16)
constexpr int PADE = 384 + 8;   // LDS row stride (elems): 4-bank skew, 16B aligned
constexpr int PADN = 256 + 8;
constexpr int PADH = 128 + 8;

// per-layer bf16 weight block offsets (elements) in workspace
constexpr size_t OFF_GE_W1 = 0;                 // [128][384]
constexpr size_t OFF_GE_W2 = 49152;             // [128][128]
constexpr size_t OFF_GE_W3 = 65536;             // [128][128]
constexpr size_t OFF_GN_W1 = 81920;             // [128][256]
constexpr size_t OFF_GN_W2 = 114688;            // [128][128]
constexpr size_t OFF_GN_W3 = 131072;            // [128][128]
constexpr size_t WSTRIDE   = 147456;            // elems per layer

// ---------- WMMA GEMM: 64 rows x 32 cols per wave, B-fragments reused over 4 M-tiles ----------
// A in LDS (bf16, row stride lda elems). W global, [128][K] row-major bf16.
template<int K>
__device__ __forceinline__ void mma_block(const unsigned short* __restrict__ a_lds, int lda,
                                          const unsigned short* __restrict__ w,
                                          int wave, int lane, v8f (&acc)[4][2]) {
  const int q = lane & 15;   // A: row m | B/C: col n
  const int h = lane >> 4;   // half selector
  const unsigned short* b0p = w + (size_t)(wave * 32 + q) * K + h * 16;
  const unsigned short* b1p = w + (size_t)(wave * 32 + 16 + q) * K + h * 16;
#pragma unroll
  for (int k0 = 0; k0 < K; k0 += 32) {
    BFrag bx, by;
    bx.w = *(const u16x16*)(b0p + k0);   // lane: 16 contiguous K for col n
    by.w = *(const u16x16*)(b1p + k0);
#pragma unroll
    for (int r = 0; r < 4; ++r) {        // reuse B across 4 row tiles
      BFrag a;
      const unsigned short* arow = a_lds + (r * 16 + q) * lda + k0;
      a.h[0] = *(const u16x8*)(arow + h * 16);       // K = k0 + h*16 .. +7
      a.h[1] = *(const u16x8*)(arow + h * 16 + 8);   // K = k0+h*16+8 .. +7
      acc[r][0] = __builtin_amdgcn_wmma_f32_16x16x32_bf16(false, a.bf, false, bx.bf,
                                                          (short)0, acc[r][0], false, false);
      acc[r][1] = __builtin_amdgcn_wmma_f32_16x16x32_bf16(false, a.bf, false, by.bf,
                                                          (short)0, acc[r][1], false, false);
    }
  }
}

__device__ __forceinline__ void zero_acc(v8f (&acc)[4][2]) {
#pragma unroll
  for (int r = 0; r < 4; ++r) {
    v8f z = {};
    acc[r][0] = z;
    acc[r][1] = z;
  }
}

// C/D layout: lane (n = lane&15, half = lane>>4); vgpr i -> row m = half*8+i (within 16-tile).
__device__ __forceinline__ void store_act_lds(unsigned short* dst, int ldd, int wave, int lane,
                                              const v8f (&acc)[4][2],
                                              const float* __restrict__ bias, bool relu) {
  const int q = lane & 15, h = lane >> 4;
  const float b0 = bias[wave * 32 + q];
  const float b1 = bias[wave * 32 + 16 + q];
#pragma unroll
  for (int r = 0; r < 4; ++r) {
#pragma unroll
    for (int i = 0; i < 8; ++i) {
      const int m = r * 16 + h * 8 + i;
      float x0 = acc[r][0][i] + b0, x1 = acc[r][1][i] + b1;
      if (relu) { x0 = fmaxf(x0, 0.f); x1 = fmaxf(x1, 0.f); }
      dst[m * ldd + wave * 32 + q]      = f32_bf16(x0);
      dst[m * ldd + wave * 32 + 16 + q] = f32_bf16(x1);
    }
  }
}

// ---------- weight transpose/convert: src [L][K][128] f32 -> dst [l][128][K] bf16 ----------
__global__ void wprep_kernel(const float* __restrict__ src, unsigned short* __restrict__ dst,
                             int K, int Ltot) {
  long long idx   = (long long)blockIdx.x * blockDim.x + threadIdx.x;
  long long total = (long long)Ltot * 128 * K;
  if (idx >= total) return;
  int k = (int)(idx % K);
  long long r = idx / K;
  int o = (int)(r % 128);
  int l = (int)(r / 128);
  dst[(size_t)l * WSTRIDE + (size_t)o * K + k] =
      f32_bf16(src[((size_t)l * K + k) * 128 + o]);
}

// ---------- GNN edge layer: e_emb = MLP3([V[s],V[r],E]); E += e_emb; agg[s] += e_emb ----------
__global__ __launch_bounds__(128) void gnn_edge_kernel(
    const float* __restrict__ V, float* __restrict__ Ebuf, const int* __restrict__ edges,
    const unsigned short* __restrict__ Wl,
    const float* __restrict__ b1, const float* __restrict__ b2, const float* __restrict__ b3,
    float* __restrict__ agg, int Etot) {
  __shared__ unsigned short xs[MT * PADE];   // input concat / later h2
  __shared__ unsigned short hs[MT * PADH];   // h1
  __shared__ int sidx[MT];
  const int tid  = threadIdx.x;
  const int wave = tid >> 5, lane = tid & 31;
  const int e0   = blockIdx.x * MT;

  // gather MT edges: [V[s] | V[r] | E] -> bf16 LDS (coalesced per row)
  for (int r = 0; r < MT; ++r) {
    const int ee = min(e0 + r, Etot - 1);
    const int si = edges[2 * ee];
    const int ri = edges[2 * ee + 1];
    xs[r * PADE + tid]       = f32_bf16(V[(size_t)si * D_ + tid]);
    xs[r * PADE + 128 + tid] = f32_bf16(V[(size_t)ri * D_ + tid]);
    xs[r * PADE + 256 + tid] = f32_bf16(Ebuf[(size_t)ee * D_ + tid]);
    if (tid == 0) sidx[r] = si;
  }
  __syncthreads();

  v8f acc[4][2];
  zero_acc(acc);
  mma_block<384>(xs, PADE, Wl + OFF_GE_W1, wave, lane, acc);
  store_act_lds(hs, PADH, wave, lane, acc, b1, true);
  __syncthreads();

  zero_acc(acc);
  mma_block<128>(hs, PADH, Wl + OFF_GE_W2, wave, lane, acc);
  store_act_lds(xs, PADH, wave, lane, acc, b2, true);   // reuse xs as h2
  __syncthreads();

  zero_acc(acc);
  mma_block<128>(xs, PADH, Wl + OFF_GE_W3, wave, lane, acc);

  const int q = lane & 15, h = lane >> 4;
  const float bb0 = b3[wave * 32 + q], bb1 = b3[wave * 32 + 16 + q];
#pragma unroll
  for (int r = 0; r < 4; ++r) {
#pragma unroll
    for (int i = 0; i < 8; ++i) {
      const int m  = r * 16 + h * 8 + i;
      const int ee = e0 + m;
      if (ee < Etot) {
        const float v0 = acc[r][0][i] + bb0;
        const float v1 = acc[r][1][i] + bb1;
        const size_t o = (size_t)ee * D_ + wave * 32 + q;
        Ebuf[o]      += v0;                 // block owns this row: plain RMW
        Ebuf[o + 16] += v1;
        const size_t s = (size_t)sidx[m] * D_ + wave * 32 + q;
        atomicAdd(&agg[s], v0);             // scatter-sum by source node
        atomicAdd(&agg[s + 16], v1);
      }
    }
  }
}

// ---------- GNN node layer: v_emb = MLP3([V, agg]); V += v_emb; agg = 0 ----------
__global__ __launch_bounds__(128) void gnn_node_kernel(
    float* __restrict__ V, float* __restrict__ agg,
    const unsigned short* __restrict__ Wl,
    const float* __restrict__ b1, const float* __restrict__ b2, const float* __restrict__ b3,
    int Ntot) {
  __shared__ unsigned short xs[MT * PADN];
  __shared__ unsigned short hs[MT * PADH];
  const int tid  = threadIdx.x;
  const int wave = tid >> 5, lane = tid & 31;
  const int n0   = blockIdx.x * MT;

  for (int r = 0; r < MT; ++r) {
    const int nd = min(n0 + r, Ntot - 1);
    const size_t o = (size_t)nd * D_ + tid;
    xs[r * PADN + tid]       = f32_bf16(V[o]);
    xs[r * PADN + 128 + tid] = f32_bf16(agg[o]);
    if (n0 + r < Ntot) agg[o] = 0.f;       // consume + reset for next layer
  }
  __syncthreads();

  v8f acc[4][2];
  zero_acc(acc);
  mma_block<256>(xs, PADN, Wl + OFF_GN_W1, wave, lane, acc);
  store_act_lds(hs, PADH, wave, lane, acc, b1, true);
  __syncthreads();

  zero_acc(acc);
  mma_block<128>(hs, PADH, Wl + OFF_GN_W2, wave, lane, acc);
  store_act_lds(xs, PADH, wave, lane, acc, b2, true);
  __syncthreads();

  zero_acc(acc);
  mma_block<128>(xs, PADH, Wl + OFF_GN_W3, wave, lane, acc);

  const int q = lane & 15, h = lane >> 4;
  const float bb0 = b3[wave * 32 + q], bb1 = b3[wave * 32 + 16 + q];
#pragma unroll
  for (int r = 0; r < 4; ++r) {
#pragma unroll
    for (int i = 0; i < 8; ++i) {
      const int m  = r * 16 + h * 8 + i;
      const int nd = n0 + m;
      if (nd < Ntot) {
        const size_t o = (size_t)nd * D_ + wave * 32 + q;
        V[o]      += acc[r][0][i] + bb0;
        V[o + 16] += acc[r][1][i] + bb1;
      }
    }
  }
}

// ---------- node encoder: MLP3(12->128->128->128) + LayerNorm ----------
__global__ __launch_bounds__(128) void enc_node_kernel(
    const float* __restrict__ pos, const float* __restrict__ areas, const float* __restrict__ info,
    const float* __restrict__ w1, const float* __restrict__ bb1,
    const float* __restrict__ w2, const float* __restrict__ bb2,
    const float* __restrict__ w3, const float* __restrict__ bb3,
    const float* __restrict__ g, const float* __restrict__ bt,
    float* __restrict__ V, int Ntot) {
  __shared__ float x[12];
  __shared__ float h1[128];
  __shared__ float h2[128];
  __shared__ float red[128];
  const int nid = blockIdx.x, t = threadIdx.x;
  if (t < 3)       x[t] = pos[(size_t)nid * 3 + t];
  else if (t == 3) x[3] = areas[nid];
  else if (t < 12) x[t] = info[t - 4];
  __syncthreads();
  float a = bb1[t];
#pragma unroll
  for (int k = 0; k < 12; ++k) a = fmaf(x[k], w1[k * 128 + t], a);
  h1[t] = fmaxf(a, 0.f);
  __syncthreads();
  a = bb2[t];
  for (int k = 0; k < 128; ++k) a = fmaf(h1[k], w2[k * 128 + t], a);
  h2[t] = fmaxf(a, 0.f);
  __syncthreads();
  a = bb3[t];
  for (int k = 0; k < 128; ++k) a = fmaf(h2[k], w3[k * 128 + t], a);
  red[t] = a; __syncthreads();
  for (int s = 64; s > 0; s >>= 1) { if (t < s) red[t] += red[t + s]; __syncthreads(); }
  const float mu = red[0] * (1.f / 128.f); __syncthreads();
  const float dv = a - mu;
  red[t] = dv * dv; __syncthreads();
  for (int s = 64; s > 0; s >>= 1) { if (t < s) red[t] += red[t + s]; __syncthreads(); }
  const float var = red[0] * (1.f / 128.f);
  V[(size_t)nid * 128 + t] = dv * rsqrtf(var + 1e-5f) * g[t] + bt[t];
}

// ---------- edge encoder: MLP3(4->128->128->128) + LayerNorm ----------
__global__ __launch_bounds__(128) void enc_edge_kernel(
    const float* __restrict__ pos, const int* __restrict__ edges,
    const float* __restrict__ w1, const float* __restrict__ bb1,
    const float* __restrict__ w2, const float* __restrict__ bb2,
    const float* __restrict__ w3, const float* __restrict__ bb3,
    const float* __restrict__ g, const float* __restrict__ bt,
    float* __restrict__ Ebuf, int Etot) {
  __shared__ float x[4];
  __shared__ float h1[128];
  __shared__ float h2[128];
  __shared__ float red[128];
  const int eid = blockIdx.x, t = threadIdx.x;
  if (t == 0) {
    const int s = edges[2 * eid], r = edges[2 * eid + 1];
    const float dx = pos[(size_t)s * 3 + 0] - pos[(size_t)r * 3 + 0];
    const float dy = pos[(size_t)s * 3 + 1] - pos[(size_t)r * 3 + 1];
    const float dz = pos[(size_t)s * 3 + 2] - pos[(size_t)r * 3 + 2];
    x[0] = dx; x[1] = dy; x[2] = dz;
    x[3] = sqrtf(dx * dx + dy * dy + dz * dz);
  }
  __syncthreads();
  float a = bb1[t];
#pragma unroll
  for (int k = 0; k < 4; ++k) a = fmaf(x[k], w1[k * 128 + t], a);
  h1[t] = fmaxf(a, 0.f);
  __syncthreads();
  a = bb2[t];
  for (int k = 0; k < 128; ++k) a = fmaf(h1[k], w2[k * 128 + t], a);
  h2[t] = fmaxf(a, 0.f);
  __syncthreads();
  a = bb3[t];
  for (int k = 0; k < 128; ++k) a = fmaf(h2[k], w3[k * 128 + t], a);
  red[t] = a; __syncthreads();
  for (int s = 64; s > 0; s >>= 1) { if (t < s) red[t] += red[t + s]; __syncthreads(); }
  const float mu = red[0] * (1.f / 128.f); __syncthreads();
  const float dv = a - mu;
  red[t] = dv * dv; __syncthreads();
  for (int s = 64; s > 0; s >>= 1) { if (t < s) red[t] += red[t + s]; __syncthreads(); }
  const float var = red[0] * (1.f / 128.f);
  Ebuf[(size_t)eid * 128 + t] = dv * rsqrtf(var + 1e-5f) * g[t] + bt[t];
}

// ---------- decoder: MLP3(128->128->128->1) ----------
__global__ __launch_bounds__(128) void dec_kernel(
    const float* __restrict__ V,
    const float* __restrict__ w1, const float* __restrict__ bb1,
    const float* __restrict__ w2, const float* __restrict__ bb2,
    const float* __restrict__ w3, const float* __restrict__ bb3,
    float* __restrict__ out, int Ntot) {
  __shared__ float h0[128];
  __shared__ float h1[128];
  __shared__ float red[128];
  const int nid = blockIdx.x, t = threadIdx.x;
  h0[t] = V[(size_t)nid * 128 + t];
  __syncthreads();
  float a = bb1[t];
  for (int k = 0; k < 128; ++k) a = fmaf(h0[k], w1[k * 128 + t], a);
  h1[t] = fmaxf(a, 0.f);
  __syncthreads();
  a = bb2[t];
  for (int k = 0; k < 128; ++k) a = fmaf(h1[k], w2[k * 128 + t], a);
  red[t] = fmaxf(a, 0.f) * w3[t];
  __syncthreads();
  for (int s = 64; s > 0; s >>= 1) { if (t < s) red[t] += red[t + s]; __syncthreads(); }
  if (t == 0) out[nid] = red[0] + bb3[0];
}

// ---------- host ----------
extern "C" void kernel_launch(void* const* d_in, const int* in_sizes, int n_in,
                              void* d_out, int out_size, void* d_ws, size_t ws_size,
                              hipStream_t stream) {
  const float* node_pos = (const float*)d_in[0];
  const float* areas    = (const float*)d_in[1];
  const float* info     = (const float*)d_in[2];
  const int*   edges    = (const int*)  d_in[3];
  const float* fv_w1 = (const float*)d_in[4],  *fv_b1 = (const float*)d_in[5];
  const float* fv_w2 = (const float*)d_in[6],  *fv_b2 = (const float*)d_in[7];
  const float* fv_w3 = (const float*)d_in[8],  *fv_b3 = (const float*)d_in[9];
  const float* fv_g  = (const float*)d_in[10], *fv_bt = (const float*)d_in[11];
  const float* fe_w1 = (const float*)d_in[12], *fe_b1 = (const float*)d_in[13];
  const float* fe_w2 = (const float*)d_in[14], *fe_b2 = (const float*)d_in[15];
  const float* fe_w3 = (const float*)d_in[16], *fe_b3 = (const float*)d_in[17];
  const float* fe_g  = (const float*)d_in[18], *fe_bt = (const float*)d_in[19];
  const float* ge_w1 = (const float*)d_in[20], *ge_b1 = (const float*)d_in[21];
  const float* ge_w2 = (const float*)d_in[22], *ge_b2 = (const float*)d_in[23];
  const float* ge_w3 = (const float*)d_in[24], *ge_b3 = (const float*)d_in[25];
  const float* gn_w1 = (const float*)d_in[26], *gn_b1 = (const float*)d_in[27];
  const float* gn_w2 = (const float*)d_in[28], *gn_b2 = (const float*)d_in[29];
  const float* gn_w3 = (const float*)d_in[30], *gn_b3 = (const float*)d_in[31];
  const float* de_w1 = (const float*)d_in[32], *de_b1 = (const float*)d_in[33];
  const float* de_w2 = (const float*)d_in[34], *de_b2 = (const float*)d_in[35];
  const float* de_w3 = (const float*)d_in[36], *de_b3 = (const float*)d_in[37];
  (void)n_in; (void)out_size; (void)ws_size;

  const int N = in_sizes[0] / 3;
  const int E = in_sizes[3] / 2;
  const int L = in_sizes[21] / 128;   // ge_b1: [L,128]

  char* ws  = (char*)d_ws;
  size_t off = 0;
  auto carve = [&](size_t bytes) {
    char* p = ws + off;
    off = (off + bytes + 255) & ~(size_t)255;
    return p;
  };
  float*          V    = (float*)carve((size_t)N * 128 * sizeof(float));
  float*          Ebuf = (float*)carve((size_t)E * 128 * sizeof(float));
  float*          agg  = (float*)carve((size_t)N * 128 * sizeof(float));
  unsigned short* Wbf  = (unsigned short*)carve((size_t)L * WSTRIDE * sizeof(unsigned short));

  auto prep = [&](const float* src, size_t dstOff, int K) {
    long long tot = (long long)L * 128 * K;
    int blocks = (int)((tot + 255) / 256);
    wprep_kernel<<<blocks, 256, 0, stream>>>(src, Wbf + dstOff, K, L);
  };
  prep(ge_w1, OFF_GE_W1, 384);
  prep(ge_w2, OFF_GE_W2, 128);
  prep(ge_w3, OFF_GE_W3, 128);
  prep(gn_w1, OFF_GN_W1, 256);
  prep(gn_w2, OFF_GN_W2, 128);
  prep(gn_w3, OFF_GN_W3, 128);

  hipMemsetAsync(agg, 0, (size_t)N * 128 * sizeof(float), stream);

  enc_node_kernel<<<N, 128, 0, stream>>>(node_pos, areas, info,
      fv_w1, fv_b1, fv_w2, fv_b2, fv_w3, fv_b3, fv_g, fv_bt, V, N);
  enc_edge_kernel<<<E, 128, 0, stream>>>(node_pos, edges,
      fe_w1, fe_b1, fe_w2, fe_b2, fe_w3, fe_b3, fe_g, fe_bt, Ebuf, E);

  for (int l = 0; l < L; ++l) {
    gnn_edge_kernel<<<(E + MT - 1) / MT, 128, 0, stream>>>(
        V, Ebuf, edges, Wbf + (size_t)l * WSTRIDE,
        ge_b1 + l * 128, ge_b2 + l * 128, ge_b3 + l * 128, agg, E);
    gnn_node_kernel<<<(N + MT - 1) / MT, 128, 0, stream>>>(
        V, agg, Wbf + (size_t)l * WSTRIDE,
        gn_b1 + l * 128, gn_b2 + l * 128, gn_b3 + l * 128, N);
  }

  dec_kernel<<<N, 128, 0, stream>>>(V, de_w1, de_b1, de_w2, de_b2, de_w3, de_b3,
                                    (float*)d_out, N);
}
